// _DummyMoeLayer_9302899163572
// MI455X (gfx1250) — matrix operand order repaired and checked by reading, not compiled, not run on hardware
//
#include <hip/hip_runtime.h>

#define N_TOK 2048
#define DIM   1024
#define IDIM  2048
#define NEXP  8

#define BM 128
#define BN 64
#define BK 32
#define LDA 40   // padded LDS stride (elements), multiple of 8 for 16B-aligned lane reads
#define LDB 40

typedef __bf16 bf16;
typedef __attribute__((ext_vector_type(8)))  __bf16 bf16x8;
typedef __attribute__((ext_vector_type(16))) __bf16 bf16x16;
typedef __attribute__((ext_vector_type(8)))  float  f32x8;
typedef int v4i_ __attribute__((__vector_size__(4 * sizeof(int))));

// ---- CDNA5 async global->LDS (ASYNCcnt) with compile-safe fallback ----
#if defined(__has_builtin)
#if __has_builtin(__builtin_amdgcn_global_load_async_to_lds_b128)
#define HAVE_ASYNC_LDS 1
#endif
#endif
#ifndef HAVE_ASYNC_LDS
#define HAVE_ASYNC_LDS 0
#endif

#if HAVE_ASYNC_LDS
#if __has_builtin(__builtin_amdgcn_s_wait_asynccnt)
#define ASYNC_WAIT() __builtin_amdgcn_s_wait_asynccnt(0)
#else
#define ASYNC_WAIT() asm volatile("s_wait_asynccnt 0x0" ::: "memory")
#endif
#else
#define ASYNC_WAIT() ((void)0)
#endif

// copy 16 bf16 (32B) global -> LDS, per thread
__device__ __forceinline__ void copy_row32(bf16* dst, const bf16* src) {
#if HAVE_ASYNC_LDS
  __builtin_amdgcn_global_load_async_to_lds_b128((v4i_*)src, (v4i_*)dst, 0, 0);
  __builtin_amdgcn_global_load_async_to_lds_b128((v4i_*)(src + 8), (v4i_*)(dst + 8), 0, 0);
#else
  bf16x8 v0 = *(const bf16x8*)src;
  bf16x8 v1 = *(const bf16x8*)(src + 8);
  *(bf16x8*)dst = v0;
  *(bf16x8*)(dst + 8) = v1;
#endif
}

__device__ __forceinline__ unsigned short bfbits(float f) {
  unsigned u = __float_as_uint(f);
  u += 0x7FFFu + ((u >> 16) & 1u);          // round-to-nearest-even
  return (unsigned short)(u >> 16);
}

__device__ __forceinline__ bf16 f2bf(float f) {
  unsigned short s = bfbits(f);
  return __builtin_bit_cast(bf16, s);
}

// Stage a BKxBN f32 weight tile transposed+converted into LDS [n][k] (packed b64 stores).
// Thread slot covers one column n and 4 consecutive k; lanes vary n -> coalesced loads.
__device__ __forceinline__ void stage_w_tile(const float* __restrict__ W, int ldw,
                                             int k0, int n0, bf16* sdst, int tid) {
#pragma unroll
  for (int it = 0; it < 2; it++) {
    int slot = it * 256 + tid;               // 512 slots = 64 n * 8 k-groups
    int n  = slot & 63;
    int kg = slot >> 6;                      // 0..7
    const float* src = W + (size_t)(k0 + kg * 4) * ldw + n0 + n;
    float a0 = src[0];
    float a1 = src[(size_t)ldw];
    float a2 = src[(size_t)2 * ldw];
    float a3 = src[(size_t)3 * ldw];
    unsigned lo = (unsigned)bfbits(a0) | ((unsigned)bfbits(a1) << 16);
    unsigned hi = (unsigned)bfbits(a2) | ((unsigned)bfbits(a3) << 16);
    unsigned long long pk = (unsigned long long)lo | ((unsigned long long)hi << 32);
    *(unsigned long long*)(sdst + n * LDB + kg * 4) = pk;   // ds_store_b64
  }
}

// -------- A fragment: 16x32 bf16, ISA 7.12.2 layout --------
__device__ __forceinline__ bf16x16 load_fragA(const bf16* base, int ld, int lane) {
  int r  = lane & 15;
  int kb = (lane >> 4) << 3;                // 0 or 8
  const bf16* p = base + r * ld + kb;
  bf16x8 lo = *(const bf16x8*)p;            // K = kb .. kb+7
  bf16x8 hi = *(const bf16x8*)(p + 16);     // K = kb+16 .. kb+23
  bf16x16 f;
#pragma unroll
  for (int i = 0; i < 8; i++) { f[i] = lo[i]; f[i + 8] = hi[i]; }
  return f;
}

// -------- B fragment: 32x16 bf16 stored column-major in LDS as [n][k] --------
__device__ __forceinline__ bf16x16 load_fragB(const bf16* base, int ld, int lane) {
  int c  = lane & 15;
  int kb = (lane >> 4) << 3;                // 0 or 8
  const bf16* p = base + c * ld + kb;
  bf16x8 lo = *(const bf16x8*)p;
  bf16x8 hi = *(const bf16x8*)(p + 16);
  bf16x16 f;
#pragma unroll
  for (int i = 0; i < 8; i++) { f[i] = lo[i]; f[i + 8] = hi[i]; }
  return f;
}

// ---------------------------------------------------------------- zero init
__global__ void zero_kernel(float* __restrict__ out, float* __restrict__ gates,
                            int* __restrict__ counts) {
  int i = blockIdx.x * blockDim.x + threadIdx.x;
  if (i < N_TOK * DIM)  out[i] = 0.f;
  if (i < N_TOK * NEXP) gates[i] = 0.f;
  if (i < NEXP)         counts[i] = 0;
}

// ---------------------------------------------------------------- x -> bf16
__global__ void xconv_kernel(const float* __restrict__ x, bf16* __restrict__ xb) {
  int i = (blockIdx.x * blockDim.x + threadIdx.x) * 8;
  bf16x8 o;
#pragma unroll
  for (int j = 0; j < 8; j++) o[j] = f2bf(x[i + j]);
  *(bf16x8*)(xb + i) = o;
}

// ---------------------------------------------------------------- router
__global__ void router_kernel(const float* __restrict__ x, const float* __restrict__ wr,
                              float* __restrict__ gates, int* __restrict__ counts,
                              int* __restrict__ ids) {
  int lane = threadIdx.x & 31;
  int n = blockIdx.x * (blockDim.x >> 5) + (threadIdx.x >> 5);
  if (n >= N_TOK) return;
  float acc[NEXP];
#pragma unroll
  for (int e = 0; e < NEXP; e++) acc[e] = 0.f;
  for (int d = lane; d < DIM; d += 32) {
    float xv = x[n * DIM + d];
    const float* wrow = wr + d * NEXP;
#pragma unroll
    for (int e = 0; e < NEXP; e++) acc[e] += xv * wrow[e];
  }
#pragma unroll
  for (int e = 0; e < NEXP; e++)
    for (int off = 16; off > 0; off >>= 1)
      acc[e] += __shfl_xor(acc[e], off, 32);
  if (lane == 0) {
    int i0 = 0; float v0 = acc[0];
    for (int e = 1; e < NEXP; e++) if (acc[e] > v0) { v0 = acc[e]; i0 = e; }
    int i1 = -1; float v1 = -3.0e38f;
    for (int e = 0; e < NEXP; e++) if (e != i0 && acc[e] > v1) { v1 = acc[e]; i1 = e; }
    float p0 = 1.f / (1.f + __expf(v1 - v0));
    float p1 = 1.f - p0;
    gates[n * NEXP + i0] = p0;
    gates[n * NEXP + i1] = p1;
    int s0 = atomicAdd(&counts[i0], 1); ids[i0 * N_TOK + s0] = n;
    int s1 = atomicAdd(&counts[i1], 1); ids[i1 * N_TOK + s1] = n;
  }
}

// ---------------------------------------------------------------- gate/up GEMM + SwiGLU
__global__ __launch_bounds__(256)
void gateup_kernel(const bf16* __restrict__ xb, const float* __restrict__ wg,
                   const float* __restrict__ wu, const int* __restrict__ counts,
                   const int* __restrict__ ids, bf16* __restrict__ hbuf) {
  int e  = blockIdx.z;
  int m0 = blockIdx.y * BM;
  int n0 = blockIdx.x * BN;
  int cnt = counts[e];
  if (m0 >= cnt) return;

  __shared__ bf16 sA[2][BM * LDA];
  __shared__ bf16 sG[2][BN * LDB];
  __shared__ bf16 sU[2][BN * LDB];

  int tid = threadIdx.x, lane = tid & 31, wid = tid >> 5;
  int wm = wid & 3, wn = wid >> 2;

  const int*   eids = ids + e * N_TOK;
  const float* gW = wg + (size_t)e * DIM * IDIM;
  const float* uW = wu + (size_t)e * DIM * IDIM;

  f32x8 accG[2][2] = {};
  f32x8 accU[2][2] = {};

  // gathered A row: clamp invalid rows to a valid token (their outputs are discarded)
  int arow = tid >> 1, ahalf = tid & 1;
  int mrow = m0 + arow; if (mrow >= cnt) mrow = cnt - 1;
  const bf16* asrc = xb + (size_t)eids[mrow] * DIM + ahalf * 16;
  int aoff = arow * LDA + ahalf * 16;

  // prologue: stage tile k0 = 0 into buffer 0
  copy_row32(&sA[0][aoff], asrc);
  stage_w_tile(gW, IDIM, 0, n0, sG[0], tid);
  stage_w_tile(uW, IDIM, 0, n0, sU[0], tid);

  int buf = 0;
  for (int k0 = 0; k0 < DIM; k0 += BK) {
    ASYNC_WAIT();
    __syncthreads();
    int nb = buf ^ 1;
    bool more = (k0 + BK) < DIM;
    if (more) copy_row32(&sA[nb][aoff], asrc + (k0 + BK));   // async -> next buffer

    bf16x16 af[2], gf[2], uf[2];
#pragma unroll
    for (int i = 0; i < 2; i++) af[i] = load_fragA(&sA[buf][(wm * 32 + i * 16) * LDA], LDA, lane);
#pragma unroll
    for (int j = 0; j < 2; j++) {
      gf[j] = load_fragB(&sG[buf][(wn * 32 + j * 16) * LDB], LDB, lane);
      uf[j] = load_fragB(&sU[buf][(wn * 32 + j * 16) * LDB], LDB, lane);
    }
    if (more) {
      stage_w_tile(gW, IDIM, k0 + BK, n0, sG[nb], tid);
      stage_w_tile(uW, IDIM, k0 + BK, n0, sU[nb], tid);
    }
#pragma unroll
    for (int i = 0; i < 2; i++)
#pragma unroll
      for (int j = 0; j < 2; j++) {
        accG[i][j] = __builtin_amdgcn_wmma_f32_16x16x32_bf16(
            false, af[i], false, gf[j], (short)0, accG[i][j], false, false);
        accU[i][j] = __builtin_amdgcn_wmma_f32_16x16x32_bf16(
            false, af[i], false, uf[j], (short)0, accU[i][j], false, false);
      }
    buf = nb;
  }

  // epilogue: h = silu(g) * u, bf16 compacted per expert
  int hi = lane >> 4, colL = lane & 15;
#pragma unroll
  for (int i = 0; i < 2; i++)
#pragma unroll
    for (int r = 0; r < 8; r++) {
      int m = m0 + wm * 32 + i * 16 + r + hi * 8;
      if (m >= cnt) continue;
      size_t rowbase = ((size_t)e * N_TOK + m) * IDIM + n0 + wn * 32;
#pragma unroll
      for (int j = 0; j < 2; j++) {
        float g = accG[i][j][r];
        float u = accU[i][j][r];
        float h = (g / (1.f + __expf(-g))) * u;
        hbuf[rowbase + j * 16 + colL] = f2bf(h);
      }
    }
}

// ---------------------------------------------------------------- down GEMM + combine
__global__ __launch_bounds__(256)
void down_kernel(const bf16* __restrict__ hbuf, const float* __restrict__ wd,
                 const int* __restrict__ counts, const int* __restrict__ ids,
                 const float* __restrict__ gates, float* __restrict__ out) {
  int e  = blockIdx.z;
  int m0 = blockIdx.y * BM;
  int n0 = blockIdx.x * BN;
  int cnt = counts[e];
  if (m0 >= cnt) return;

  __shared__ bf16 sA[2][BM * LDA];
  __shared__ bf16 sB[2][BN * LDB];

  int tid = threadIdx.x, lane = tid & 31, wid = tid >> 5;
  int wm = wid & 3, wn = wid >> 2;

  const bf16*  hbase = hbuf + (size_t)e * N_TOK * IDIM;
  const float* dW    = wd + (size_t)e * IDIM * DIM;

  f32x8 acc[2][2] = {};
  int arow = tid >> 1, ahalf = tid & 1;
  const bf16* asrc = hbase + (size_t)(m0 + arow) * IDIM + ahalf * 16;
  int aoff = arow * LDA + ahalf * 16;

  copy_row32(&sA[0][aoff], asrc);
  stage_w_tile(dW, DIM, 0, n0, sB[0], tid);

  int buf = 0;
  for (int k0 = 0; k0 < IDIM; k0 += BK) {
    ASYNC_WAIT();
    __syncthreads();
    int nb = buf ^ 1;
    bool more = (k0 + BK) < IDIM;
    if (more) copy_row32(&sA[nb][aoff], asrc + (k0 + BK));

    bf16x16 af[2], bfr[2];
#pragma unroll
    for (int i = 0; i < 2; i++) af[i]  = load_fragA(&sA[buf][(wm * 32 + i * 16) * LDA], LDA, lane);
#pragma unroll
    for (int j = 0; j < 2; j++) bfr[j] = load_fragB(&sB[buf][(wn * 32 + j * 16) * LDB], LDB, lane);

    if (more) stage_w_tile(dW, DIM, k0 + BK, n0, sB[nb], tid);

#pragma unroll
    for (int i = 0; i < 2; i++)
#pragma unroll
      for (int j = 0; j < 2; j++)
        acc[i][j] = __builtin_amdgcn_wmma_f32_16x16x32_bf16(
            false, af[i], false, bfr[j], (short)0, acc[i][j], false, false);
    buf = nb;
  }

  int hi = lane >> 4, colL = lane & 15;
#pragma unroll
  for (int i = 0; i < 2; i++)
#pragma unroll
    for (int r = 0; r < 8; r++) {
      int m = m0 + wm * 32 + i * 16 + r + hi * 8;
      if (m >= cnt) continue;
      int tok = ids[e * N_TOK + m];
      float p = gates[tok * NEXP + e];
#pragma unroll
      for (int j = 0; j < 2; j++) {
        int col = n0 + wn * 32 + j * 16 + colL;
        atomicAdd(&out[(size_t)tok * DIM + col], p * acc[i][j][r]);
      }
    }
}

// ----------------------------------------------------------------
extern "C" void kernel_launch(void* const* d_in, const int* in_sizes, int n_in,
                              void* d_out, int out_size, void* d_ws, size_t ws_size,
                              hipStream_t stream) {
  const float* x  = (const float*)d_in[0];
  const float* wr = (const float*)d_in[1];
  const float* wg = (const float*)d_in[2];
  const float* wu = (const float*)d_in[3];
  const float* wd = (const float*)d_in[4];
  float* out = (float*)d_out;

  char* ws = (char*)d_ws;
  float* gates  = (float*)ws;  ws += (size_t)N_TOK * NEXP * sizeof(float);
  int*   counts = (int*)ws;    ws += 256;
  int*   ids    = (int*)ws;    ws += (size_t)NEXP * N_TOK * sizeof(int);
  bf16*  xb     = (bf16*)ws;   ws += (size_t)N_TOK * DIM * sizeof(bf16);
  bf16*  hbuf   = (bf16*)ws;   // E * N * I * 2 bytes

  zero_kernel<<<(N_TOK * DIM + 255) / 256, 256, 0, stream>>>(out, gates, counts);
  xconv_kernel<<<(N_TOK * DIM / 8 + 255) / 256, 256, 0, stream>>>(x, xb);
  router_kernel<<<N_TOK / 8, 256, 0, stream>>>(x, wr, gates, counts, ids);

  dim3 g1(IDIM / BN, N_TOK / BM, NEXP);
  gateup_kernel<<<g1, 256, 0, stream>>>(xb, wg, wu, counts, ids, hbuf);

  dim3 g2(DIM / BN, N_TOK / BM, NEXP);
  down_kernel<<<g2, 256, 0, stream>>>(hbuf, wd, counts, ids, gates, out);
}
